// TransLayer_188978561044
// MI455X (gfx1250) — matrix operand order repaired and checked by reading, not compiled, not run on hardware
//
#include <hip/hip_runtime.h>
#include <stdint.h>

typedef unsigned short u16;
typedef unsigned int   u32;
typedef unsigned long long u64;
typedef __attribute__((ext_vector_type(16))) __bf16 v16bf;
typedef __attribute__((ext_vector_type(8)))  float  v8f;
typedef __attribute__((ext_vector_type(2)))  float  v2f;
typedef __attribute__((ext_vector_type(4)))  unsigned int u32x4;
typedef __attribute__((ext_vector_type(8)))  int i32x8;
typedef __attribute__((ext_vector_type(4)))  int i32x4;

#define DEVINL __device__ __forceinline__

#if defined(__has_builtin)
# if __has_builtin(__builtin_amdgcn_tensor_load_to_lds) && __has_builtin(__builtin_amdgcn_s_wait_tensorcnt)
#  define USE_TDM 1
# endif
#endif

DEVINL u16 f2bf(float f){ union{float f; u32 u;} x; x.f=f; u32 r = x.u + 0x7FFFu + ((x.u>>16)&1u); return (u16)(r>>16); }
DEVINL float bf2f(u16 s){ union{u32 u; float f;} x; x.u = ((u32)s)<<16; return x.f; }

union FragBF { v16bf v; u16 s[16]; u32 u[8]; };

DEVINL v8f vzero8(){ v8f z;
#pragma unroll
  for (int i=0;i<8;++i) z[i]=0.f;
  return z; }

// ---------------- WMMA fragment loaders (CDNA5 wave32 layouts) ----------------
// All loaders read contiguous-along-K 32-bit words so they merge into b64/b128.
// A fragment: 16x32 bf16 (row-major src). lane m=lane&15, kg=lane>>4.
// VGPR j holds K = 8*kg + 2*(j&3) + 16*(j>>2) (+1)
DEVINL v16bf load_a_bf(const u16* Arow, int k0, int kg){
  FragBF f;
#pragma unroll
  for (int j=0;j<8;++j){
    int k = k0 + 8*kg + ((j&3)<<1) + ((j>>2)<<4);
    f.u[j] = *(const u32*)(Arow + k);
  }
  return f.v;
}
// B fragment where the source is N-major (Bt[N][K]); Brow = &Bt[n][0].
// VGPR j holds K = 8*kg + 2*(j&3) + 16*(j>>2) (+1) (same as A).
DEVINL v16bf load_b_bf_t(const u16* Brow, int k0, int kg){
  FragBF f;
#pragma unroll
  for (int j=0;j<8;++j){
    int k = k0 + 8*kg + ((j&3)<<1) + ((j>>2)<<4);
    f.u[j] = *(const u32*)(Brow + k);
  }
  return f.v;
}
DEVINL v8f wmma_bf16(v16bf a, v16bf b, v8f c){
  return __builtin_amdgcn_wmma_f32_16x16x32_bf16(false, a, false, b, (short)0, c, false, false);
}

// ---------------- Tensor Data Mover: 2D tile -> LDS ----------------
#ifdef USE_TDM
DEVINL void tdm_load_2d(u32 lds_off, const void* gptr, u32 dsz,
                        u32 tile0, u32 tile1, u32 stride0){
  u64 ga = (u64)(size_t)gptr;
  u32x4 g0 = { 1u,                                  // count=1, user descriptor
               lds_off,                             // LDS byte address
               (u32)ga,                             // global_addr[31:0]
               (u32)(ga>>32) | (2u<<30) };          // global_addr[56:32] | type=2
  i32x8 g1 = { (int)(dsz<<16),                      // wg_mask=0, data_size
               (int)((tile0&0xFFFFu)<<16),          // tensor_dim0[15:0] (=tile0)
               (int)((tile0>>16) | ((tile1&0xFFFFu)<<16)),   // dim0 hi | dim1 lo
               (int)((tile1>>16) | ((tile0&0xFFFFu)<<16)),   // dim1 hi | tile_dim0
               (int)(tile1&0xFFFFu),                // tile_dim1 | tile_dim2=0
               (int)stride0, 0, 0 };                // tensor_dim0_stride
  i32x4 z4 = {0,0,0,0};
#if defined(__clang_major__) && (__clang_major__ >= 23)
  i32x8 z8 = {0,0,0,0,0,0,0,0};
  __builtin_amdgcn_tensor_load_to_lds(g0, g1, z4, z4, z8, 0);
#else
  __builtin_amdgcn_tensor_load_to_lds(g0, g1, z4, z4, 0);
#endif
}
#endif

DEVINL void stage_bf16(u16* lds, const u16* g, int tile0, int tile1, int strideElems, int wv){
#ifdef USE_TDM
  if (wv==0) tdm_load_2d((u32)(size_t)lds, g, 1u, (u32)tile0, (u32)tile1, (u32)strideElems);
#else
  (void)wv;
  for (int i = threadIdx.x; i < tile0*tile1; i += blockDim.x){
    int r = i / tile0, c = i - r*tile0;
    lds[i] = g[(size_t)r*strideElems + c];
  }
#endif
}
DEVINL void stage_f32(float* lds, const float* g, int tile0, int tile1, int strideElems, int wv){
#ifdef USE_TDM
  if (wv==0) tdm_load_2d((u32)(size_t)lds, g, 2u, (u32)tile0, (u32)tile1, (u32)strideElems);
#else
  (void)wv;
  for (int i = threadIdx.x; i < tile0*tile1; i += blockDim.x){
    int r = i / tile0, c = i - r*tile0;
    lds[i] = g[(size_t)r*strideElems + c];
  }
#endif
}
DEVINL void stage_wait(int wv){
#ifdef USE_TDM
  if (wv==0) __builtin_amdgcn_s_wait_tensorcnt(0);
#else
  (void)wv;
#endif
}

// ---------------------------------------------------------------- kernels

// Convert weights to bf16 *transposed* (N-major) so all GEMM B-fragments are
// contiguous-along-K: WQT[1536][512], WOT[512][512]^T.
__global__ void k_cvt_weights(const float* __restrict__ wqkv, const float* __restrict__ wout,
                              u16* __restrict__ wq_t, u16* __restrict__ wo_t){
  int i = blockIdx.x*256 + threadIdx.x;
  if (i < 512*1536){ int k = i/1536, c = i%1536; wq_t[(size_t)c*512 + k] = f2bf(wqkv[i]); }
  if (i < 512*512) { int k = i/512,  c = i%512;  wo_t[(size_t)c*512 + k] = f2bf(wout[i]); }
}

__global__ void k_layernorm(const float* __restrict__ x, const float* __restrict__ w,
                            const float* __restrict__ b, u16* __restrict__ xn){
  int row  = blockIdx.x*8 + (threadIdx.x>>5);
  int lane = threadIdx.x & 31;
  const float* xr = x + (size_t)row*512;
  float v[16]; float s=0.f, s2=0.f;
#pragma unroll
  for (int i=0;i<16;++i){ v[i]=xr[lane + 32*i]; s+=v[i]; s2+=v[i]*v[i]; }
#pragma unroll
  for (int m=16;m>=1;m>>=1){ s += __shfl_xor(s,m,32); s2 += __shfl_xor(s2,m,32); }
  float mu  = s*(1.f/512.f);
  float var = s2*(1.f/512.f) - mu*mu;
  float rs  = rsqrtf(var + 1e-5f);
  u16* o = xn + (size_t)row*512;
#pragma unroll
  for (int i=0;i<16;++i){ int c = lane+32*i; o[c] = f2bf((v[i]-mu)*rs*w[c] + b[c]); }
}

// qkv = xn[16384,512] @ wqkv[512,1536]; TDM-staged A stripe + double-buffered
// transposed-B tiles [128n x 32k]. Also emits v transposed (VBT) for flash P@V.
__global__ void k_gemm_qkv(const u16* __restrict__ A, const u16* __restrict__ Bw /*WQT*/,
                           u16* __restrict__ qb, u16* __restrict__ kb,
                           u16* __restrict__ vb, u16* __restrict__ vbt){
  __shared__ __align__(16) u16 At[16*512];      // 16KB
  __shared__ __align__(16) u16 Bt[2][128*32];   // 2 x 8KB
  int wv = threadIdx.x>>5, lane = threadIdx.x&31, m = lane&15, kg = lane>>4;
  int tile = blockIdx.x*8 + wv;
  int ti = tile/96, tj = tile%96;
  int tj0 = (blockIdx.x*8)%96;
  stage_bf16(At, A + (size_t)ti*16*512, 512, 16, 512, wv);
  stage_bf16(Bt[0], Bw + (size_t)tj0*16*512, 32, 128, 512, wv);
  stage_wait(wv);
  __syncthreads();
  v8f acc = vzero8();
  for (int c=0; c<16; ++c){                   // K chunks of 32
    int cur = c&1;
    if (c+1 < 16) stage_bf16(Bt[cur^1], Bw + (size_t)tj0*16*512 + (c+1)*32, 32, 128, 512, wv);
    acc = wmma_bf16(load_a_bf(At + m*512, c*32, kg),
                    load_b_bf_t(Bt[cur] + (wv*16 + m)*32, 0, kg), acc);
    if (c+1 < 16) stage_wait(wv);
    __syncthreads();
  }
#pragma unroll
  for (int r=0;r<8;++r){
    int row = ti*16 + r + 8*kg;
    int col = tj*16 + m;
    int which = col>>9, hc = col&511, h = hc>>6, d = hc&63;
    int bb = row>>12, n = row&4095;
    int bh = bb*8+h;
    size_t o = ((size_t)bh*4096 + n)*64 + d;
    float vv = acc[r];
    if      (which==0) qb[o] = f2bf(vv*0.125f);   // DH^-0.5
    else if (which==1) kb[o] = f2bf(vv);
    else { u16 bfv = f2bf(vv); vb[o] = bfv;
           vbt[(size_t)bh*262144 + (size_t)d*4096 + n] = bfv; }
  }
}

__global__ void k_landmarks(const u16* __restrict__ qb, const u16* __restrict__ kb,
                            u16* __restrict__ ql, u16* __restrict__ kl){
  int i = blockIdx.x*256 + threadIdx.x;
  const int half = 32*256*64;
  const u16* src = (i<half) ? qb : kb;
  u16*       dst = (i<half) ? ql : kl;
  int e  = (i<half) ? i : i-half;
  int bh = e/(256*64), rem = e%(256*64), mi = rem>>6, d = rem&63;
  const u16* p = src + ((size_t)bh*4096 + mi*16)*64 + d;
  float s = 0.f;
#pragma unroll
  for (int t=0;t<16;++t) s += bf2f(p[t*64]);
  dst[e] = f2bf(s*(1.f/16.f));
}

// sim2 = q_l @ k_l^T (pre-softmax), fp32 out
__global__ void k_sim2(const u16* __restrict__ ql, const u16* __restrict__ kl, float* __restrict__ a2){
  int wv = threadIdx.x>>5, lane=threadIdx.x&31, m=lane&15, kg=lane>>4;
  int tile = blockIdx.x*8 + wv;
  int bh = tile>>8, t = tile&255, ti=t>>4, tj=t&15;
  const u16* Arow = ql + ((size_t)bh*256 + ti*16 + m)*64;
  const u16* Brow = kl + ((size_t)bh*256 + tj*16 + m)*64;
  v8f acc = vzero8();
  acc = wmma_bf16(load_a_bf(Arow, 0,kg), load_b_bf_t(Brow, 0,kg), acc);
  acc = wmma_bf16(load_a_bf(Arow,32,kg), load_b_bf_t(Brow,32,kg), acc);
  float* out = a2 + (size_t)bh*65536;
#pragma unroll
  for (int r=0;r<8;++r) out[(ti*16 + r + 8*kg)*256 + tj*16 + m] = acc[r];
}

__global__ void k_softmax256(float* __restrict__ a){
  int row  = blockIdx.x*8 + (threadIdx.x>>5);
  int lane = threadIdx.x&31;
  float* p = a + (size_t)row*256;
  float v[8]; float mx=-1e30f;
#pragma unroll
  for (int i=0;i<8;++i){ v[i]=p[lane+32*i]; mx=fmaxf(mx,v[i]); }
#pragma unroll
  for (int m=16;m>=1;m>>=1) mx = fmaxf(mx, __shfl_xor(mx,m,32));
  float s=0.f;
#pragma unroll
  for (int i=0;i<8;++i){ v[i]=__expf(v[i]-mx); s+=v[i]; }
#pragma unroll
  for (int m=16;m>=1;m>>=1) s += __shfl_xor(s,m,32);
  float inv = 1.f/s;
#pragma unroll
  for (int i=0;i<8;++i) p[lane+32*i] = v[i]*inv;
}

__global__ void k_init_scalar(u32* s){ if (threadIdx.x==0 && blockIdx.x==0) *s = 0u; }

__global__ void k_colsum_max(const float* __restrict__ a2, u32* s){
  int bh = blockIdx.x, col = threadIdx.x;
  const float* p = a2 + (size_t)bh*65536 + col;
  float sum=0.f;
  for (int i=0;i<256;++i) sum += p[i*256];
  union{float f; u32 u;} x; x.f=sum;
  atomicMax(s, x.u);
}

// z0 = a2^T / (max_rowsum(=1) * max_colsum)
__global__ void k_z0(const float* __restrict__ a2, const u32* __restrict__ s, float* __restrict__ z){
  int i = blockIdx.x*256 + threadIdx.x;
  union{u32 u; float f;} x; x.u = *s;
  float inv = 1.f/x.f;
  int bh = i>>16, rem = i&65535, r = rem>>8, c = rem&255;
  z[i] = a2[(size_t)bh*65536 + c*256 + r]*inv;
}

// D = alpha * A @ (cdiag*I + sgn*B), batched 32 x [256,256,256], fp32 WMMA.
__global__ void k_pinv_gemm(const float* __restrict__ A, const float* __restrict__ Bm,
                            float* __restrict__ D, float cdiag, float sgn, float alpha){
  __shared__ __align__(16) float Af[16*256];
  __shared__ __align__(16) float Bf[2][32*128];
  int bh = blockIdx.y;
  int wv = threadIdx.x>>5, lane=threadIdx.x&31, m=lane&15, kg=lane>>4;
  int tile = blockIdx.x*8 + wv;
  int ti = tile>>4, tj = tile&15;
  int colbase = (blockIdx.x&1)*128;
  const float* Ab = A  + (size_t)bh*65536;
  const float* Bb = Bm + (size_t)bh*65536;
  int ncol = tj*16 + m;
  int cloc = wv*16 + m;
  stage_f32(Af, Ab + (size_t)ti*16*256, 256, 16, 256, wv);
  stage_f32(Bf[0], Bb + colbase, 128, 32, 256, wv);
  stage_wait(wv);
  __syncthreads();
  v8f acc = vzero8();
  for (int c=0; c<8; ++c){                   // K chunks of 32
    int cur = c&1;
    if (c+1 < 8) stage_f32(Bf[cur^1], Bb + (size_t)(c+1)*32*256 + colbase, 128, 32, 256, wv);
    const float* Arow = Af + m*256 + c*32;
    const float* Bc   = Bf[cur];
#pragma unroll
    for (int kk=0; kk<32; kk+=4){
      int kl = kk + 2*kg;
      int kglob = c*32 + kl;
      v2f a, b;
      a[0] = Arow[kl];   a[1] = Arow[kl+1];
      b[0] = sgn*Bc[kl*128 + cloc]     + ((kglob   == ncol) ? cdiag : 0.f);
      b[1] = sgn*Bc[(kl+1)*128 + cloc] + ((kglob+1 == ncol) ? cdiag : 0.f);
      acc = __builtin_amdgcn_wmma_f32_16x16x4_f32(false, a, false, b, (short)0, acc, false, false);
    }
    if (c+1 < 8) stage_wait(wv);
    __syncthreads();
  }
  float* Db = D + (size_t)bh*65536;
#pragma unroll
  for (int r=0;r<8;++r) Db[(ti*16 + r + 8*kg)*256 + tj*16 + m] = alpha*acc[r];
}

__global__ void k_cvt_f32_bf(const float* __restrict__ src, u16* __restrict__ dst, int n){
  int i = blockIdx.x*256 + threadIdx.x;
  if (i<n) dst[i] = f2bf(src[i]);
}

// flash: Out = softmax(Q @ K^T) @ V per batch-head.
// K rows ([kRows,64]) and V *transposed* ([64,kRows]) chunks TDM-staged in LDS.
// writeBf=1 -> write output transposed [64, qRows] (bf16); else [qRows,64] fp32.
__global__ void k_flash(const u16* __restrict__ Q, const u16* __restrict__ Kd,
                        const u16* __restrict__ Vt, float* __restrict__ Of,
                        u16* __restrict__ Obf, int qRows, int kRows, int writeBf){
  __shared__ __align__(16) u16 Ktl[2][32*64];     // 2 x 4KB (rows of K)
  __shared__ __align__(16) u16 Vtl[2][64*32];     // 2 x 4KB (V^T: [d][k])
  __shared__ __align__(16) u16 plds[4][512];      // per-wave P re-layout bounce
  int wv = threadIdx.x>>5, lane = threadIdx.x&31, m = lane&15, kg = lane>>4;
  int gw = blockIdx.x*4 + wv;
  int rbPerBatch = qRows>>4;
  int bh = gw / rbPerBatch, rb = gw % rbPerBatch, row0 = rb<<4;
  const u16* Qrow = Q  + ((size_t)bh*qRows + row0 + m)*64;
  const u16* Kb   = Kd + (size_t)bh*kRows*64;
  const u16* Vb   = Vt + (size_t)bh*64*kRows;     // [64][kRows]
  v16bf aq0 = load_a_bf(Qrow, 0, kg);
  v16bf aq1 = load_a_bf(Qrow,32, kg);
  v8f o[4];
#pragma unroll
  for (int t=0;t<4;++t) o[t]=vzero8();
  float mrow[8], lrow[8];
#pragma unroll
  for (int r=0;r<8;++r){ mrow[r]=-1e30f; lrow[r]=0.f; }
  u16* pl = plds[wv];
  int nch = kRows>>5;
  stage_bf16(Ktl[0], Kb, 64, 32, 64, wv);
  stage_bf16(Vtl[0], Vb, 32, 64, kRows, wv);
  stage_wait(wv);
  __syncthreads();

  for (int c=0; c<nch; ++c){
    int cur = c&1;
    if (c+1 < nch){
      stage_bf16(Ktl[cur^1], Kb + (size_t)(c+1)*32*64, 64, 32, 64, wv);
      stage_bf16(Vtl[cur^1], Vb + (size_t)(c+1)*32,    32, 64, kRows, wv);
    }
    const u16* Kc = Ktl[cur];
    const u16* Vc = Vtl[cur];
    v8f s0=vzero8(), s1=vzero8();
    s0 = wmma_bf16(aq0, load_b_bf_t(Kc + m*64,      0, kg), s0);
    s0 = wmma_bf16(aq1, load_b_bf_t(Kc + m*64,     32, kg), s0);
    s1 = wmma_bf16(aq0, load_b_bf_t(Kc + (16+m)*64, 0, kg), s1);
    s1 = wmma_bf16(aq1, load_b_bf_t(Kc + (16+m)*64,32, kg), s1);
    float cm[8];
#pragma unroll
    for (int r=0;r<8;++r) cm[r] = fmaxf(s0[r], s1[r]);
#pragma unroll
    for (int msk=8;msk>=1;msk>>=1){
#pragma unroll
      for (int r=0;r<8;++r) cm[r] = fmaxf(cm[r], __shfl_xor(cm[r], msk, 32));
    }
    float ps[8];
#pragma unroll
    for (int r=0;r<8;++r){
      float mn = fmaxf(mrow[r], cm[r]);
      float sc = __expf(mrow[r]-mn);
      mrow[r]=mn; lrow[r]*=sc;
#pragma unroll
      for (int t=0;t<4;++t) o[t][r]*=sc;
      float p0 = __expf(s0[r]-mn), p1 = __expf(s1[r]-mn);
      s0[r]=p0; s1[r]=p1; ps[r]=p0+p1;
    }
#pragma unroll
    for (int msk=8;msk>=1;msk>>=1){
#pragma unroll
      for (int r=0;r<8;++r) ps[r] += __shfl_xor(ps[r], msk, 32);
    }
#pragma unroll
    for (int r=0;r<8;++r) lrow[r]+=ps[r];
#pragma unroll
    for (int r=0;r<8;++r){
      pl[(r+8*kg)*32 + m]      = f2bf(s0[r]);
      pl[(r+8*kg)*32 + m + 16] = f2bf(s1[r]);
    }
    v16bf pa = load_a_bf(pl + m*32, 0, kg);
#pragma unroll
    for (int t=0;t<4;++t)
      o[t] = wmma_bf16(pa, load_b_bf_t(Vc + (t*16 + m)*32, 0, kg), o[t]);
    if (c+1 < nch) stage_wait(wv);
    __syncthreads();
  }
#pragma unroll
  for (int r=0;r<8;++r){
    float inv = 1.f/lrow[r];
    int row = row0 + r + 8*kg;
#pragma unroll
    for (int t=0;t<4;++t){
      float val = o[t][r]*inv;
      if (writeBf) Obf[(size_t)bh*64*qRows + (size_t)(t*16+m)*qRows + row] = f2bf(val);
      else         Of [((size_t)bh*qRows + row)*64 + t*16 + m] = val;
    }
  }
}

// w2 = z(bf16)[256,256] @ a3v[256,64]; B read from a3v^T [64][256];
// output written transposed w2t [64][256] (it feeds flash as V).
__global__ void k_w2(const u16* __restrict__ zb, const u16* __restrict__ a3vt, u16* __restrict__ w2t){
  int wv=threadIdx.x>>5, lane=threadIdx.x&31, m=lane&15, kg=lane>>4;
  int tile = blockIdx.x*8+wv;
  int bh = tile>>6, t=tile&63, ti=t>>2, tj=t&3;
  const u16* Arow = zb   + (size_t)bh*65536 + (ti*16+m)*256;
  const u16* Brow = a3vt + (size_t)bh*16384 + (tj*16+m)*256;   // [d][k]
  v8f acc=vzero8();
#pragma unroll
  for (int k0=0;k0<256;k0+=32)
    acc = wmma_bf16(load_a_bf(Arow,k0,kg), load_b_bf_t(Brow,k0,kg), acc);
  u16* out = w2t + (size_t)bh*16384;
#pragma unroll
  for (int r=0;r<8;++r) out[(size_t)(tj*16+m)*256 + ti*16 + r + 8*kg] = f2bf(acc[r]);
}

// out_heads += depthwise conv(v) over seq, then merge heads -> [B*N, 512] bf16
__global__ void k_conv_combine(const float* __restrict__ attn, const u16* __restrict__ vb,
                               const float* __restrict__ rw, u16* __restrict__ out){
  int i = blockIdx.x*256 + threadIdx.x;
  int d = i&63, n = (i>>6)&4095, bh = i>>18;
  int b = bh>>3, h = bh&7;
  float val = attn[i];
  const u16* vp = vb + ((size_t)bh*4096)*64 + d;
  const float* w = rw + h*33;
#pragma unroll
  for (int t=0;t<33;++t){
    int tok = n + t - 16;
    if (tok>=0 && tok<4096) val += bf2f(vp[(size_t)tok*64]) * w[t];
  }
  out[(size_t)(b*4096+n)*512 + h*64 + d] = f2bf(val);
}

// y = attn[16384,512] @ wout[512,512] + b_out + x (fp32). B from WOT (transposed).
__global__ void k_final(const u16* __restrict__ A, const u16* __restrict__ Bw /*WOT*/,
                        const float* __restrict__ bout, const float* __restrict__ x,
                        float* __restrict__ y){
  __shared__ __align__(16) u16 At[16*512];
  __shared__ __align__(16) u16 Bt[2][128*32];
  int wv=threadIdx.x>>5, lane=threadIdx.x&31, m=lane&15, kg=lane>>4;
  int tile = blockIdx.x*8+wv;
  int ti=tile>>5, tj=tile&31;
  int tj0 = (blockIdx.x*8)&31;
  stage_bf16(At, A + (size_t)ti*16*512, 512, 16, 512, wv);
  stage_bf16(Bt[0], Bw + (size_t)tj0*16*512, 32, 128, 512, wv);
  stage_wait(wv);
  __syncthreads();
  v8f acc=vzero8();
  for (int c=0;c<16;++c){
    int cur = c&1;
    if (c+1 < 16) stage_bf16(Bt[cur^1], Bw + (size_t)tj0*16*512 + (c+1)*32, 32, 128, 512, wv);
    acc = wmma_bf16(load_a_bf(At + m*512, c*32, kg),
                    load_b_bf_t(Bt[cur] + (wv*16 + m)*32, 0, kg), acc);
    if (c+1 < 16) stage_wait(wv);
    __syncthreads();
  }
#pragma unroll
  for (int r=0;r<8;++r){
    int row = ti*16+r+8*kg, col = tj*16+m;
    size_t o = (size_t)row*512+col;
    y[o] = acc[r] + bout[col] + x[o];
  }
}

// ---------------------------------------------------------------- launch

extern "C" void kernel_launch(void* const* d_in, const int* in_sizes, int n_in,
                              void* d_out, int out_size, void* d_ws, size_t ws_size,
                              hipStream_t stream){
  (void)in_sizes; (void)n_in; (void)out_size; (void)ws_size;
  const float* x    = (const float*)d_in[0];
  const float* nw   = (const float*)d_in[1];
  const float* nb   = (const float*)d_in[2];
  const float* wqkv = (const float*)d_in[3];
  const float* wout = (const float*)d_in[4];
  const float* bout = (const float*)d_in[5];
  const float* resw = (const float*)d_in[6];
  float* y = (float*)d_out;
  char* ws = (char*)d_ws;

  size_t off = 0;
  auto alloc = [&](size_t bytes)->size_t{ size_t o = off; off += (bytes+255)&~(size_t)255; return o; };
  const size_t XN  = alloc((size_t)16384*512*2);
  const size_t WQT = alloc((size_t)1536*512*2);
  const size_t WOT = alloc((size_t)512*512*2);
  const size_t QB  = alloc((size_t)32*4096*64*2);
  const size_t KB  = alloc((size_t)32*4096*64*2);
  const size_t VB  = alloc((size_t)32*4096*64*2);
  const size_t VBT = alloc((size_t)32*64*4096*2);
  const size_t QL  = alloc((size_t)32*256*64*2);
  const size_t KL  = alloc((size_t)32*256*64*2);
  const size_t A2  = alloc((size_t)32*256*256*4);   // attn_f32 aliases A2..T5 after pinv
  const size_t T1  = alloc((size_t)32*256*256*4);
  const size_t T3  = alloc((size_t)32*256*256*4);
  const size_t T5  = alloc((size_t)32*256*256*4);
  const size_t Z   = alloc((size_t)32*256*256*4);
  const size_t Z2  = alloc((size_t)32*256*256*4);
  const size_t ZB  = alloc((size_t)32*256*256*2);
  const size_t A3VT= alloc((size_t)32*64*256*2);
  const size_t W2T = alloc((size_t)32*64*256*2);
  const size_t ABF = alloc((size_t)16384*512*2);
  const size_t SCL = alloc(4);

  u16*   XNp  = (u16*)(ws+XN);
  u16*   WQTp = (u16*)(ws+WQT);
  u16*   WOTp = (u16*)(ws+WOT);
  u16*   QBp  = (u16*)(ws+QB);
  u16*   KBp  = (u16*)(ws+KB);
  u16*   VBp  = (u16*)(ws+VB);
  u16*   VBTp = (u16*)(ws+VBT);
  u16*   QLp  = (u16*)(ws+QL);
  u16*   KLp  = (u16*)(ws+KL);
  float* A2p  = (float*)(ws+A2);
  float* T1p  = (float*)(ws+T1);
  float* T3p  = (float*)(ws+T3);
  float* T5p  = (float*)(ws+T5);
  float* Zp   = (float*)(ws+Z);
  float* Z2p  = (float*)(ws+Z2);
  u16*   ZBp  = (u16*)(ws+ZB);
  u16*   A3VTp= (u16*)(ws+A3VT);
  u16*   W2Tp = (u16*)(ws+W2T);
  u16*   ABFp = (u16*)(ws+ABF);
  u32*   SCLp = (u32*)(ws+SCL);
  float* ATTNFp = (float*)(ws+A2);                  // alias (pinv temps dead by then)

  k_cvt_weights<<<3072,256,0,stream>>>(wqkv, wout, WQTp, WOTp);
  k_layernorm  <<<2048,256,0,stream>>>(x, nw, nb, XNp);
  k_gemm_qkv   <<<12288,256,0,stream>>>(XNp, WQTp, QBp, KBp, VBp, VBTp);
  k_landmarks  <<<4096,256,0,stream>>>(QBp, KBp, QLp, KLp);
  k_sim2       <<<1024,256,0,stream>>>(QLp, KLp, A2p);
  k_softmax256 <<<1024,256,0,stream>>>(A2p);
  k_init_scalar<<<1,32,0,stream>>>(SCLp);
  k_colsum_max <<<32,256,0,stream>>>(A2p, SCLp);
  k_z0         <<<8192,256,0,stream>>>(A2p, SCLp, Zp);

  float* cur = Zp; float* nxt = Z2p;
  for (int it=0; it<6; ++it){
    k_pinv_gemm<<<dim3(32,32),256,0,stream>>>(A2p, cur, T1p,  0.f,  1.f, 1.f );  // t1 = x@z
    k_pinv_gemm<<<dim3(32,32),256,0,stream>>>(T1p, T1p, T3p,  7.f, -1.f, 1.f );  // t3 = t1@(7I-t1)
    k_pinv_gemm<<<dim3(32,32),256,0,stream>>>(T1p, T3p, T5p, 15.f, -1.f, 1.f );  // t5 = t1@(15I-t3)
    k_pinv_gemm<<<dim3(32,32),256,0,stream>>>(cur, T5p, nxt, 13.f, -1.f, 0.25f); // z' = .25 z@(13I-t5)
    float* tmp = cur; cur = nxt; nxt = tmp;
  }
  k_cvt_f32_bf <<<8192,256,0,stream>>>(cur, ZBp, 32*256*256);

  // a3v^T = (softmax(q_l @ k^T) @ v)^T    [bh,64,256] bf16
  k_flash<<<128,128,0,stream>>>(QLp, KBp, VBTp, nullptr, A3VTp, 256, 4096, 1);
  // w2^T = (a2inv @ a3v)^T                [bh,64,256] bf16
  k_w2   <<<256,256,0,stream>>>(ZBp, A3VTp, W2Tp);
  // out_heads = softmax(q @ k_l^T) @ w2   [bh,4096,64] fp32 (aliases pinv temps)
  k_flash<<<2048,128,0,stream>>>(QBp, KLp, W2Tp, ATTNFp, nullptr, 4096, 256, 0);

  k_conv_combine<<<32768,256,0,stream>>>(ATTNFp, VBp, resw, ABFp);
  k_final       <<<4096,256,0,stream>>>(ABFp, WOTp, bout, x, y);
}
